// MultiHeadAttention_73667279061265
// MI455X (gfx1250) — matrix operand order repaired and checked by reading, not compiled, not run on hardware
//
#include <hip/hip_runtime.h>
#include <hip/hip_bf16.h>

// ---------------------------------------------------------------------------
// Multi-head attention (B=4, S=2048, D=1024, H=16, d_k=64) for gfx1250.
// bf16 WMMA (v_wmma_f32_16x16x32_bf16) everywhere, f32 accumulation,
// flash-attention online softmax in f32.  Tile staging uses CDNA5
// GLOBAL_LOAD_ASYNC_TO_LDS_B128 with double-buffered LDS (ASYNCcnt).
// ---------------------------------------------------------------------------

typedef __attribute__((ext_vector_type(16))) __bf16          v16bf;
typedef __attribute__((ext_vector_type(8)))  float           v8f;
typedef __attribute__((ext_vector_type(8)))  unsigned short  u16x8;

union FragU { v16bf bf; u16x8 u[2]; };

__device__ __forceinline__ unsigned short f32_to_bf16(float x) {
  union { float f; unsigned int u; } c; c.f = x;
  unsigned int r = c.u + 0x7FFFu + ((c.u >> 16) & 1u);   // round to nearest even
  return (unsigned short)(r >> 16);
}

__device__ __forceinline__ v8f wmma_bf16(v16bf a, v16bf b, v8f c) {
  return __builtin_amdgcn_wmma_f32_16x16x32_bf16(false, a, false, b, (short)0, c,
                                                 false, false);
}

// Low 32 bits of a flat shared-memory address == wave-relative LDS byte offset.
__device__ __forceinline__ unsigned lds_off(const void* p) {
  return (unsigned)(unsigned long long)p;
}

// Issue one 16B/lane async copy global->LDS (tracked by ASYNCcnt).
__device__ __forceinline__ void async_b128(unsigned ldsByte, const void* gaddr) {
  asm volatile("global_load_async_to_lds_b128 %0, %1, off"
               :: "v"(ldsByte), "v"((unsigned long long)gaddr) : "memory");
}

__device__ __forceinline__ void wait_async0() {
  asm volatile("s_wait_asynccnt 0x0" ::: "memory");
}

// A-matrix fragment (16x32 bf16): lane<16 holds row lane, K={0..7,16..23};
// lane>=16 holds row lane-16, K={8..15,24..31}.  p points at halfword data.
__device__ __forceinline__ v16bf load_fragA(const unsigned short* p, int rowBase,
                                            int pitch, int lane, int kOff) {
  const int half8 = (lane & 16) ? 8 : 0;
  const unsigned short* b = p + (rowBase + (lane & 15)) * pitch + kOff + half8;
  FragU f;
  f.u[0] = *(const u16x8*)(b);
  f.u[1] = *(const u16x8*)(b + 16);
  return f.bf;
}

// B-matrix fragment (32x16 bf16), source tile stored as [N][K] row-major:
// lane<16 holds column lane, K={0..7,16..23}; lane>=16 holds column lane-16, K={8..15,24..31}.
__device__ __forceinline__ v16bf load_fragB(const unsigned short* p, int colBase,
                                            int pitch, int lane, int kOff) {
  const int half16 = (lane & 16) ? 8 : 0;
  const unsigned short* b = p + (colBase + (lane & 15)) * pitch + kOff + half16;
  FragU f;
  f.u[0] = *(const u16x8*)(b);
  f.u[1] = *(const u16x8*)(b + 16);
  return f.bf;
}

// ---------------------------------------------------------------------------
// Kernel 0: convert query/key/value fp32 -> bf16 (read once, GEMM rereads bf16)
// ---------------------------------------------------------------------------
__global__ __launch_bounds__(256) void convert_x_bf16(
    const float* __restrict__ q, const float* __restrict__ k,
    const float* __restrict__ v, unsigned short* __restrict__ xq,
    unsigned short* __restrict__ xk, unsigned short* __restrict__ xv) {
  const float* src = (blockIdx.y == 0) ? q : (blockIdx.y == 1) ? k : v;
  unsigned short* dst = (blockIdx.y == 0) ? xq : (blockIdx.y == 1) ? xk : xv;
  size_t base = ((size_t)blockIdx.x * 256 + threadIdx.x) * 8;
  float4 a = *(const float4*)(src + base);
  float4 b = *(const float4*)(src + base + 4);
  u16x8 o;
  o[0] = f32_to_bf16(a.x); o[1] = f32_to_bf16(a.y);
  o[2] = f32_to_bf16(a.z); o[3] = f32_to_bf16(a.w);
  o[4] = f32_to_bf16(b.x); o[5] = f32_to_bf16(b.y);
  o[6] = f32_to_bf16(b.z); o[7] = f32_to_bf16(b.w);
  *(u16x8*)(dst + base) = o;
}

// ---------------------------------------------------------------------------
// Kernel 1: transpose + convert the four 1024x1024 fp32 weights into bf16 W^T
// (output layout [N][K]), concatenated: Wq^T | Wk^T | Wv^T | Wo^T.
// ---------------------------------------------------------------------------
__global__ void transpose_w_bf16(const float* __restrict__ Wq,
                                 const float* __restrict__ Wk,
                                 const float* __restrict__ Wv,
                                 const float* __restrict__ Wo,
                                 unsigned short* __restrict__ WT) {
  __shared__ float tile[32][33];
  const float* W = (blockIdx.z == 0) ? Wq : (blockIdx.z == 1) ? Wk
                 : (blockIdx.z == 2) ? Wv : Wo;
  unsigned short* out = WT + (size_t)blockIdx.z * 1024u * 1024u;
  const int tx = threadIdx.x, ty = threadIdx.y;
  const int n = blockIdx.x * 32 + tx;
#pragma unroll
  for (int j = 0; j < 32; j += 8) {
    int k = blockIdx.y * 32 + ty + j;
    tile[ty + j][tx] = W[k * 1024 + n];
  }
  __syncthreads();
  const int kk = blockIdx.y * 32 + tx;
#pragma unroll
  for (int j = 0; j < 32; j += 8) {
    int nn = blockIdx.x * 32 + ty + j;
    out[nn * 1024 + kk] = f32_to_bf16(tile[tx][ty + j]);
  }
}

// ---------------------------------------------------------------------------
// Kernel 2: 8192x1024x1024 GEMM,  C = A(bf16) * W^T(bf16 [N][K]) + bias.
//   mode 0: out bf16 * 0.125 -> Q  [b,h,s,d]
//   mode 1: out bf16         -> K  [b,h,s,d]
//   mode 2: out bf16         -> V^T[b,h,d,s]
//   mode 3: out fp32         -> flat [8192][1024]
// Block tile 128x128 (4 waves, 64x64/wave), k-step 32, async double-buffered.
// ---------------------------------------------------------------------------
__global__ __launch_bounds__(128) void gemm128(const unsigned short* __restrict__ A,
                                               const unsigned short* __restrict__ WT,
                                               const float* __restrict__ bias,
                                               unsigned short* __restrict__ outU,
                                               float* __restrict__ outF,
                                               int mode) {
  // pitch 40 halfwords (80B) per row; two buffers each for A and B tiles
  __shared__ __align__(16) unsigned short lsA[2 * 128 * 40];
  __shared__ __align__(16) unsigned short lsB[2 * 128 * 40];

  const int tid  = threadIdx.x;
  const int lane = tid & 31;
  const int wave = tid >> 5;
  const int m0 = blockIdx.y * 128;
  const int n0 = blockIdx.x * 128;
  const int mW = (wave >> 1) * 64;
  const int nW = (wave & 1) * 64;
  const unsigned aOff = lds_off(lsA);
  const unsigned bOff = lds_off(lsB);

  auto issue_tile = [&](int kk, int buf) {
    const unsigned aBase = aOff + buf * 10240;   // 128*40*2 bytes
    const unsigned bBase = bOff + buf * 10240;
#pragma unroll
    for (int i = 0; i < 4; ++i) {
      int flat = i * 128 + tid;
      int r = flat >> 2, c = flat & 3;
      async_b128(aBase + r * 80 + c * 16, A  + (size_t)(m0 + r) * 1024 + kk + c * 8);
      async_b128(bBase + r * 80 + c * 16, WT + (size_t)(n0 + r) * 1024 + kk + c * 8);
    }
  };

  v8f zero = {};
  v8f acc[4][4];
#pragma unroll
  for (int i = 0; i < 4; ++i)
#pragma unroll
    for (int j = 0; j < 4; ++j) acc[i][j] = zero;

  issue_tile(0, 0);
  int cur = 0;
  for (int k0 = 0; k0 < 1024; k0 += 32) {
    wait_async0();
    __syncthreads();
    if (k0 + 32 < 1024) issue_tile(k0 + 32, cur ^ 1);

    const unsigned short* lA = lsA + cur * 5120;   // halfwords
    const unsigned short* lB = lsB + cur * 5120;
    v16bf af[4], bf[4];
#pragma unroll
    for (int t = 0; t < 4; ++t) af[t] = load_fragA(lA, mW + t * 16, 40, lane, 0);
#pragma unroll
    for (int t = 0; t < 4; ++t) bf[t] = load_fragB(lB, nW + t * 16, 40, lane, 0);
#pragma unroll
    for (int mt = 0; mt < 4; ++mt)
#pragma unroll
      for (int nt = 0; nt < 4; ++nt)
        acc[mt][nt] = wmma_bf16(af[mt], bf[nt], acc[mt][nt]);
    cur ^= 1;
  }

  // ---- epilogue ----
  const int halfAdd = (lane & 16) ? 8 : 0;
#pragma unroll
  for (int mt = 0; mt < 4; ++mt) {
#pragma unroll
    for (int nt = 0; nt < 4; ++nt) {
      int col = n0 + nW + nt * 16 + (lane & 15);
      float bv = bias[col];
#pragma unroll
      for (int g = 0; g < 8; ++g) {
        int row = m0 + mW + mt * 16 + g + halfAdd;
        float v = acc[mt][nt][g] + bv;
        if (mode == 3) {
          outF[(size_t)row * 1024 + col] = v;
        } else {
          int bi = row >> 11, s = row & 2047;
          int h  = col >> 6,  d = col & 63;
          if (mode == 0) v *= 0.125f;            // fold 1/sqrt(d_k) into Q
          unsigned short o = f32_to_bf16(v);
          if (mode == 2)
            outU[(size_t)((bi * 16 + h) * 64 + d) * 2048 + s] = o;   // V^T
          else
            outU[(size_t)((bi * 16 + h) * 2048 + s) * 64 + d] = o;   // Q / K
        }
      }
    }
  }
}

// ---------------------------------------------------------------------------
// Kernel 3: flash attention per (b,h).  Workgroup = 128 threads (4 waves),
// 128 queries per workgroup (32 per wave), key blocks of 128, d_k = 64.
// Q pre-scaled by 1/8.  K/V^T tiles double-buffered via async-to-LDS.
// Output O as bf16 [b*2048+s][h*64+d] (8192x1024).
// ---------------------------------------------------------------------------
__global__ __launch_bounds__(128) void attn_flash(const unsigned short* __restrict__ Q,
                                                  const unsigned short* __restrict__ Kb,
                                                  const unsigned short* __restrict__ VT,
                                                  unsigned short* __restrict__ O) {
  __shared__ __align__(16) unsigned short lsK[2 * 128 * 72];   // [key][d]  pitch 72
  __shared__ __align__(16) unsigned short lsV[2 * 64 * 136];   // [d][key]  pitch 136
  __shared__ __align__(16) unsigned short lsP[4 * 32 * 136];   // per-wave P tiles

  const int tid  = threadIdx.x;
  const int lane = tid & 31;
  const int wave = tid >> 5;
  const int halfAdd = (lane & 16) ? 8 : 0;

  const int bh = blockIdx.y;                     // 0..63
  const int bb = bh >> 4, hh = bh & 15;
  const int qW = blockIdx.x * 128 + wave * 32;   // first query row of this wave

  const unsigned short* Qg = Q  + (size_t)bh * 2048 * 64;
  const unsigned short* Kg = Kb + (size_t)bh * 2048 * 64;
  const unsigned short* Vg = VT + (size_t)bh * 64 * 2048;
  unsigned short* lsPw = lsP + wave * 32 * 136;
  const unsigned kOffB = lds_off(lsK);
  const unsigned vOffB = lds_off(lsV);

  auto issue_kv = [&](int jj, int buf) {
    const int kb2 = jj * 128;
    const unsigned kBase = kOffB + buf * 18432;  // 128*72*2 bytes
    const unsigned vBase = vOffB + buf * 17408;  // 64*136*2 bytes
#pragma unroll
    for (int i = 0; i < 8; ++i) {
      int flat = i * 128 + tid;
      int r = flat >> 3, c = flat & 7;
      async_b128(kBase + r * 144 + c * 16, Kg + (size_t)(kb2 + r) * 64 + c * 8);
    }
#pragma unroll
    for (int i = 0; i < 8; ++i) {
      int flat = i * 128 + tid;
      int r = flat >> 4, c = flat & 15;
      async_b128(vBase + r * 272 + c * 16, Vg + (size_t)r * 2048 + kb2 + c * 8);
    }
  };

  // Preload Q fragments: 2 M-tiles x 2 k-steps (d_k = 64)
  v16bf qf[2][2];
#pragma unroll
  for (int mt = 0; mt < 2; ++mt)
#pragma unroll
    for (int ks = 0; ks < 2; ++ks) {
      const unsigned short* p =
          Qg + (size_t)(qW + mt * 16 + (lane & 15)) * 64 + ks * 32 + halfAdd;
      FragU f;
      f.u[0] = *(const u16x8*)(p);
      f.u[1] = *(const u16x8*)(p + 16);
      qf[mt][ks] = f.bf;
    }

  float rmax[2][8], rsum[2][8];
  v8f zero = {};
  v8f oac[2][4];
#pragma unroll
  for (int mt = 0; mt < 2; ++mt) {
#pragma unroll
    for (int g = 0; g < 8; ++g) { rmax[mt][g] = -__builtin_inff(); rsum[mt][g] = 0.f; }
#pragma unroll
    for (int n = 0; n < 4; ++n) oac[mt][n] = zero;
  }

  issue_kv(0, 0);
  int cur = 0;
  for (int j = 0; j < 16; ++j) {
    wait_async0();
    __syncthreads();
    if (j + 1 < 16) issue_kv(j + 1, cur ^ 1);

    const unsigned short* lK = lsK + cur * 9216;   // halfwords
    const unsigned short* lV = lsV + cur * 8704;

    // ---- S = Q * K^T  (scaled already) : 32 WMMAs ----
    v8f sc[2][8];
#pragma unroll
    for (int mt = 0; mt < 2; ++mt)
#pragma unroll
      for (int nt = 0; nt < 8; ++nt) sc[mt][nt] = zero;
#pragma unroll
    for (int ks = 0; ks < 2; ++ks)
#pragma unroll
      for (int nt = 0; nt < 8; ++nt) {
        v16bf kf = load_fragB(lK, nt * 16, 72, lane, ks * 32);
#pragma unroll
        for (int mt = 0; mt < 2; ++mt)
          sc[mt][nt] = wmma_bf16(qf[mt][ks], kf, sc[mt][nt]);
      }

    // ---- online softmax, write P (bf16) to per-wave LDS ----
#pragma unroll
    for (int mt = 0; mt < 2; ++mt) {
      float mb[8];
#pragma unroll
      for (int g = 0; g < 8; ++g) {
        float m = sc[mt][0][g];
#pragma unroll
        for (int nt = 1; nt < 8; ++nt) m = fmaxf(m, sc[mt][nt][g]);
        mb[g] = m;
      }
#pragma unroll
      for (int off = 8; off >= 1; off >>= 1)
#pragma unroll
        for (int g = 0; g < 8; ++g)
          mb[g] = fmaxf(mb[g], __shfl_xor(mb[g], off, 32));
      float alpha[8];
#pragma unroll
      for (int g = 0; g < 8; ++g) {
        float mnew = fmaxf(rmax[mt][g], mb[g]);
        alpha[g] = __expf(rmax[mt][g] - mnew);
        rmax[mt][g] = mnew;
      }
      float rs[8];
#pragma unroll
      for (int g = 0; g < 8; ++g) rs[g] = 0.f;
#pragma unroll
      for (int nt = 0; nt < 8; ++nt)
#pragma unroll
        for (int g = 0; g < 8; ++g) {
          float p = __expf(sc[mt][nt][g] - rmax[mt][g]);
          rs[g] += p;
          lsPw[(mt * 16 + g + halfAdd) * 136 + nt * 16 + (lane & 15)] =
              f32_to_bf16(p);
        }
#pragma unroll
      for (int off = 8; off >= 1; off >>= 1)
#pragma unroll
        for (int g = 0; g < 8; ++g) rs[g] += __shfl_xor(rs[g], off, 32);
#pragma unroll
      for (int g = 0; g < 8; ++g) rsum[mt][g] = rsum[mt][g] * alpha[g] + rs[g];
#pragma unroll
      for (int n = 0; n < 4; ++n)
#pragma unroll
        for (int g = 0; g < 8; ++g) oac[mt][n][g] *= alpha[g];
    }

    // ---- O += P * V : 32 WMMAs ----
#pragma unroll
    for (int ks = 0; ks < 4; ++ks) {
      v16bf pa[2];
#pragma unroll
      for (int mt = 0; mt < 2; ++mt)
        pa[mt] = load_fragA(lsPw, mt * 16, 136, lane, ks * 32);
#pragma unroll
      for (int n = 0; n < 4; ++n) {
        v16bf vf = load_fragB(lV, n * 16, 136, lane, ks * 32);
#pragma unroll
        for (int mt = 0; mt < 2; ++mt)
          oac[mt][n] = wmma_bf16(pa[mt], vf, oac[mt][n]);
      }
    }
    cur ^= 1;
  }

  // ---- normalize and store O as bf16 [b*2048+s][h*64+d] ----
#pragma unroll
  for (int mt = 0; mt < 2; ++mt) {
    float inv[8];
#pragma unroll
    for (int g = 0; g < 8; ++g) inv[g] = 1.0f / rsum[mt][g];
#pragma unroll
    for (int n = 0; n < 4; ++n) {
      int d = n * 16 + (lane & 15);
#pragma unroll
      for (int g = 0; g < 8; ++g) {
        int s = qW + mt * 16 + g + halfAdd;
        O[(size_t)(bb * 2048 + s) * 1024 + hh * 64 + d] =
            f32_to_bf16(oac[mt][n][g] * inv[g]);
      }
    }
  }
}

// ---------------------------------------------------------------------------
// Launch: ws layout (bytes):
//   [0,8M)    W^T bf16 x4        [8M,24M)   Q bf16 [b,h,s,d]
//   [24M,40M) K bf16 [b,h,s,d]   [40M,56M)  V^T bf16 [b,h,d,s]
//   [56M,72M) O bf16 [8192][1024]
//   [72M,88M) Xq bf16  [88M,104M) Xk bf16  [104M,120M) Xv bf16
// ---------------------------------------------------------------------------
extern "C" void kernel_launch(void* const* d_in, const int* in_sizes, int n_in,
                              void* d_out, int out_size, void* d_ws, size_t ws_size,
                              hipStream_t stream) {
  const float* query = (const float*)d_in[0];
  const float* key   = (const float*)d_in[1];
  const float* value = (const float*)d_in[2];
  const float* Wq = (const float*)d_in[3];
  const float* bq = (const float*)d_in[4];
  const float* Wk = (const float*)d_in[5];
  const float* bk = (const float*)d_in[6];
  const float* Wv = (const float*)d_in[7];
  const float* bv = (const float*)d_in[8];
  const float* Wo = (const float*)d_in[9];
  const float* bo = (const float*)d_in[10];

  char* ws = (char*)d_ws;
  unsigned short* WT  = (unsigned short*)(ws);
  unsigned short* Qb  = (unsigned short*)(ws + ((size_t)8   << 20));
  unsigned short* Kbf = (unsigned short*)(ws + ((size_t)24  << 20));
  unsigned short* VTb = (unsigned short*)(ws + ((size_t)40  << 20));
  unsigned short* Ob  = (unsigned short*)(ws + ((size_t)56  << 20));
  unsigned short* Xq  = (unsigned short*)(ws + ((size_t)72  << 20));
  unsigned short* Xk  = (unsigned short*)(ws + ((size_t)88  << 20));
  unsigned short* Xv  = (unsigned short*)(ws + ((size_t)104 << 20));
  float* out = (float*)d_out;

  convert_x_bf16<<<dim3(4096, 3), 256, 0, stream>>>(query, key, value, Xq, Xk, Xv);
  transpose_w_bf16<<<dim3(32, 32, 4), dim3(32, 8), 0, stream>>>(Wq, Wk, Wv, Wo, WT);
  gemm128<<<dim3(8, 64), 128, 0, stream>>>(Xq, WT + 0u * 1048576u, bq, Qb,  nullptr, 0);
  gemm128<<<dim3(8, 64), 128, 0, stream>>>(Xk, WT + 1u * 1048576u, bk, Kbf, nullptr, 1);
  gemm128<<<dim3(8, 64), 128, 0, stream>>>(Xv, WT + 2u * 1048576u, bv, VTb, nullptr, 2);
  attn_flash<<<dim3(16, 64), 128, 0, stream>>>(Qb, Kbf, VTb, Ob);
  gemm128<<<dim3(8, 64), 128, 0, stream>>>(Ob, WT + 3u * 1048576u, bo, nullptr, out, 3);
}
